// HierarchicalLoss_83476984365300
// MI455X (gfx1250) — compile-verified, matrix operand order checked by reading, not run on hardware
//
#include <hip/hip_runtime.h>
#include <math.h>

#define BSZ   8192
#define DDIM  256
#define NW    (BSZ / 32)   // mask words per row = 256
#define TEMP  0.07f
#define TH1   0.5f
#define TH2   0.3f
#define ALPHA 0.5f
#define LDSW  72   // bf16 row stride (halfwords): 144B rows -> 16 disjoint 4-bank groups for b128 reads

typedef __attribute__((ext_vector_type(8)))  float  v8f;
typedef __attribute__((ext_vector_type(8)))  __bf16 v8bf;
typedef __attribute__((ext_vector_type(16))) __bf16 v16bf;

// Load one 16-bit A/B WMMA fragment from row-major bf16 LDS.
// Per ISA layout: VGPR0-3 = K[8h .. 8h+7], VGPR4-7 = K[16+8h .. 16+8h+7]
__device__ __forceinline__ v16bf ldfrag(const __bf16* p) {
    v8bf a = *(const v8bf*)(p);
    v8bf b = *(const v8bf*)(p + 16);
    return __builtin_shufflevector(a, b, 0, 1, 2, 3, 4, 5, 6, 7,
                                   8, 9, 10, 11, 12, 13, 14, 15);
}

// ------- kernel 0: L2 normalize (f32-exact) + bf16 hi/lo split --------------
__global__ __launch_bounds__(DDIM) void l2norm_split_kernel(
    const float* __restrict__ in, __bf16* __restrict__ hi, __bf16* __restrict__ lo) {
    const int row = blockIdx.x;
    const int tid = threadIdx.x;
    float x = in[(size_t)row * DDIM + tid];
    float v = x * x;
#pragma unroll
    for (int m = 16; m >= 1; m >>= 1) v += __shfl_xor(v, m, 32);
    __shared__ float wsum[8];
    __shared__ float inv_norm;
    if ((tid & 31) == 0) wsum[tid >> 5] = v;
    __syncthreads();
    if (tid == 0) {
        float s = 0.f;
        for (int i = 0; i < 8; ++i) s += wsum[i];
        inv_norm = 1.0f / sqrtf(s);
    }
    __syncthreads();
    float xn = x * inv_norm;
    __bf16 h = (__bf16)xn;                    // RNE
    __bf16 l = (__bf16)(xn - (float)h);       // residual: ~16-17 effective mantissa bits
    hi[(size_t)row * DDIM + tid] = h;
    lo[(size_t)row * DDIM + tid] = l;
}

// ------- kernel 1: one sweep of scores -> bit masks + exact row counts ------
// negbits[row*NW+w] bit b  <->  scores[row][w*32+b] <= TH2
// posbits[row*NW+w] bit b  <->  scores[row][w*32+b] >  TH1
__global__ __launch_bounds__(256) void mask_kernel(
    const float* __restrict__ scores,
    unsigned* __restrict__ negbits, unsigned* __restrict__ posbits,
    float* __restrict__ n_neg, float* __restrict__ n_pos) {
    const int row  = blockIdx.x;
    const int tid  = threadIdx.x;
    const int lane = tid & 31;
    const int wv   = tid >> 5;
    int cn = 0, cp = 0;
#pragma unroll 4
    for (int it = 0; it < 32; ++it) {
        int c0 = wv * 32 + it * 256;
        float s = scores[(size_t)row * BSZ + c0 + lane];
        unsigned nm = (unsigned)__ballot(s <= TH2);
        unsigned pm = (unsigned)__ballot(s > TH1);
        if (lane == 0) {
            negbits[(size_t)row * NW + (c0 >> 5)] = nm;
            posbits[(size_t)row * NW + (c0 >> 5)] = pm;
            cn += __popc(nm);
            cp += __popc(pm);
        }
    }
    __shared__ int cns[8], cps[8];
    if (lane == 0) { cns[wv] = cn; cps[wv] = cp; }
    __syncthreads();
    if (tid == 0) {
        int tn = 0, tp = 0;
        for (int i = 0; i < 8; ++i) { tn += cns[i]; tp += cps[i]; }
        n_neg[row] = (float)tn;
        n_pos[row] = (float)tp;
    }
}

// ------- fused bf16x3 GEMM + masked-logsumexp tile kernel -------------------
// PASS 1 (maskbits = negbits): S_i2t[m] += sum_neg exp(L[m,n]); S_t2i[n] += sum_neg exp(L[m,n])
// PASS 2 (maskbits = posbits): P_i2t[m] += sum_pos log1p(S_i2t[m]*exp(-L[m,n]));
//                              P_t2i[n] += sum_pos log1p(S_t2i[n]*exp(-L[m,n]))
template <int PASS>
__global__ __launch_bounds__(256) void tile_kernel(
    const __bf16* __restrict__ fiH, const __bf16* __restrict__ fiL,
    const __bf16* __restrict__ ftH, const __bf16* __restrict__ ftL,
    const unsigned* __restrict__ maskbits,
    float* __restrict__ S_i2t, float* __restrict__ S_t2i,
    float* __restrict__ P_i2t, float* __restrict__ P_t2i) {
    __shared__ __align__(16) unsigned char smem[4 * 64 * LDSW * 2];  // 36864B
    __bf16* lAh = (__bf16*)(smem);
    __bf16* lAl = (__bf16*)(smem + 1 * 64 * LDSW * 2);
    __bf16* lBh = (__bf16*)(smem + 2 * 64 * LDSW * 2);
    __bf16* lBl = (__bf16*)(smem + 3 * 64 * LDSW * 2);

    const int i0 = blockIdx.y * 64;       // logits row block (i2t rows)
    const int j0 = blockIdx.x * 64;       // logits col block (t2i rows)
    const int tid  = threadIdx.x;
    const int wave = tid >> 5;
    const int lane = tid & 31;
    const int h    = lane >> 4;
    const int ln   = lane & 15;
    const int waveRow = wave >> 1;        // 0..3
    const int waveCol = wave & 1;         // 0..1

    v8f acc0 = {};
    v8f acc1 = {};

    const int aRow  = (waveRow * 16 + ln) * LDSW;
    const int b0Row = (waveCol * 32 + ln) * LDSW;
    const int b1Row = (waveCol * 32 + 16 + ln) * LDSW;

    for (int kc = 0; kc < DDIM; kc += 64) {
        __syncthreads();
#pragma unroll
        for (int it = 0; it < 2; ++it) {        // 512 b128 groups / 256 threads
            int flat = tid + it * 256;
            int r  = flat >> 3;
            int c8 = flat & 7;
            size_t ga = (size_t)(i0 + r) * DDIM + kc + c8 * 8;
            size_t gb = (size_t)(j0 + r) * DDIM + kc + c8 * 8;
            int lo_off = r * LDSW + c8 * 8;
            *(int4*)(lAh + lo_off) = *(const int4*)(fiH + ga);
            *(int4*)(lAl + lo_off) = *(const int4*)(fiL + ga);
            *(int4*)(lBh + lo_off) = *(const int4*)(ftH + gb);
            *(int4*)(lBl + lo_off) = *(const int4*)(ftL + gb);
        }
        __syncthreads();
#pragma unroll
        for (int k0 = 0; k0 < 64; k0 += 32) {
            const int base = k0 + 8 * h;
            v16bf aH  = ldfrag(lAh + aRow + base);
            v16bf aL  = ldfrag(lAl + aRow + base);
            v16bf b0H = ldfrag(lBh + b0Row + base);
            v16bf b0L = ldfrag(lBl + b0Row + base);
            v16bf b1H = ldfrag(lBh + b1Row + base);
            v16bf b1L = ldfrag(lBl + b1Row + base);
            // bf16x3: hi*hi + hi*lo + lo*hi (lo*lo dropped, ~2^-18 relative)
            acc0 = __builtin_amdgcn_wmma_f32_16x16x32_bf16(false, aH, false, b0H, (short)0, acc0, false, false);
            acc0 = __builtin_amdgcn_wmma_f32_16x16x32_bf16(false, aH, false, b0L, (short)0, acc0, false, false);
            acc0 = __builtin_amdgcn_wmma_f32_16x16x32_bf16(false, aL, false, b0H, (short)0, acc0, false, false);
            acc1 = __builtin_amdgcn_wmma_f32_16x16x32_bf16(false, aH, false, b1H, (short)0, acc1, false, false);
            acc1 = __builtin_amdgcn_wmma_f32_16x16x32_bf16(false, aH, false, b1L, (short)0, acc1, false, false);
            acc1 = __builtin_amdgcn_wmma_f32_16x16x32_bf16(false, aL, false, b1H, (short)0, acc1, false, false);
        }
    }

    // ---------------- stage mask bit-tiles (reuse LDS) ----------------------
    __syncthreads();
    unsigned* mIJ = (unsigned*)smem;          // [64][2] words: rows i0+, bit-cols j0+
    unsigned* mJI = (unsigned*)smem + 128;    // [64][2] words: rows j0+, bit-cols i0+
    if (tid < 256) {
        int tt = tid & 127;
        int r = tt >> 1, w = tt & 1;
        unsigned bitsw = (tid < 128)
            ? maskbits[(size_t)(i0 + r) * NW + (j0 >> 5) + w]
            : maskbits[(size_t)(j0 + r) * NW + (i0 >> 5) + w];
        if (tid < 128) mIJ[tt] = bitsw; else mJI[tt] = bitsw;
    }
    __syncthreads();

    const float invT = 1.0f / TEMP;
    float rowVal[8];
#pragma unroll
    for (int v = 0; v < 8; ++v) rowVal[v] = 0.f;

    float Srow[8];
    if (PASS == 2) {
#pragma unroll
        for (int v = 0; v < 8; ++v)
            Srow[v] = S_i2t[i0 + waveRow * 16 + v + 8 * h];
    }

#pragma unroll
    for (int t = 0; t < 2; ++t) {
        v8f c = (t == 0) ? acc0 : acc1;
        const int nloc = waveCol * 32 + t * 16 + ln;
        const int nbit = nloc & 31;
        const unsigned wJI = mJI[nloc * 2 + (waveRow >> 1)];  // word for rows mloc of this wave
        float Scol = 0.f;
        if (PASS == 2) Scol = S_t2i[j0 + nloc];
        float colAcc = 0.f;
#pragma unroll
        for (int v = 0; v < 8; ++v) {
            const int mloc = waveRow * 16 + v + 8 * h;
            float x = c[v] * invT;
            const unsigned wIJ = mIJ[mloc * 2 + waveCol];     // 16-lane broadcast
            bool mi = (wIJ >> nbit) & 1u;
            bool mt = (wJI >> (mloc & 31)) & 1u;
            if (PASS == 1) {
                float e = __expf(x);
                rowVal[v] += mi ? e : 0.f;
                colAcc    += mt ? e : 0.f;
            } else {
                float em = __expf(-x);
                rowVal[v] += mi ? log1pf(Srow[v] * em) : 0.f;
                colAcc    += mt ? log1pf(Scol * em) : 0.f;
            }
        }
        colAcc += __shfl_xor(colAcc, 16, 32);       // both half-waves hold same column
        if (h == 0) {
            if (PASS == 1) atomicAdd(&S_t2i[j0 + nloc], colAcc);
            else           atomicAdd(&P_t2i[j0 + nloc], colAcc);
        }
    }

    // one butterfly per row (accumulated over both column tiles)
#pragma unroll
    for (int v = 0; v < 8; ++v) {
        float rv = rowVal[v];
#pragma unroll
        for (int m = 8; m >= 1; m >>= 1) rv += __shfl_xor(rv, m, 32);
        if (ln == v) {
            const int mrow = i0 + waveRow * 16 + v + 8 * h;
            if (PASS == 1) atomicAdd(&S_i2t[mrow], rv);
            else           atomicAdd(&P_i2t[mrow], rv);
        }
    }
}

// ---------------- final scalar reduction ------------------------------------
__global__ __launch_bounds__(256) void finalize_kernel(
    const float* __restrict__ n_neg, const float* __restrict__ n_pos,
    const float* __restrict__ P_i2t, const float* __restrict__ P_t2i,
    float* __restrict__ out) {
    const int tid = threadIdx.x;
    float acc = 0.f;
    for (int l = tid; l < BSZ; l += 256) {
        float w = 1.0f / ((1.0f + n_neg[l]) * n_pos[l]);
        acc += (ALPHA * P_i2t[l] + (1.0f - ALPHA) * P_t2i[l]) * w;
    }
#pragma unroll
    for (int m = 16; m >= 1; m >>= 1) acc += __shfl_xor(acc, m, 32);
    __shared__ float wsum[8];
    if ((tid & 31) == 0) wsum[tid >> 5] = acc;
    __syncthreads();
    if (tid == 0) {
        float tot = 0.f;
        for (int i = 0; i < 8; ++i) tot += wsum[i];
        out[0] = tot / (float)BSZ;
    }
}

// ---------------- host launcher ---------------------------------------------
extern "C" void kernel_launch(void* const* d_in, const int* in_sizes, int n_in,
                              void* d_out, int out_size, void* d_ws, size_t ws_size,
                              hipStream_t stream) {
    (void)in_sizes; (void)n_in; (void)out_size; (void)ws_size;
    const float* f_img  = (const float*)d_in[0];
    const float* f_txt  = (const float*)d_in[1];
    const float* scores = (const float*)d_in[2];
    float* out = (float*)d_out;

    const size_t nfeat = (size_t)BSZ * DDIM;
    const size_t nmask = (size_t)BSZ * NW;
    __bf16* fiH = (__bf16*)d_ws;
    __bf16* fiL = fiH + nfeat;
    __bf16* ftH = fiL + nfeat;
    __bf16* ftL = ftH + nfeat;
    unsigned* negbits = (unsigned*)(ftL + nfeat);
    unsigned* posbits = negbits + nmask;
    float* S_i2t = (float*)(posbits + nmask);
    float* S_t2i = S_i2t + BSZ;
    float* P_i2t = S_t2i + BSZ;
    float* P_t2i = P_i2t + BSZ;
    float* nneg  = P_t2i + BSZ;
    float* npos  = nneg + BSZ;

    hipMemsetAsync(S_i2t, 0, (size_t)4 * BSZ * sizeof(float), stream);  // S and P arrays

    l2norm_split_kernel<<<BSZ, DDIM, 0, stream>>>(f_img, fiH, fiL);
    l2norm_split_kernel<<<BSZ, DDIM, 0, stream>>>(f_txt, ftH, ftL);
    mask_kernel<<<BSZ, 256, 0, stream>>>(scores, negbits, posbits, nneg, npos);

    dim3 grid(BSZ / 64, BSZ / 64);
    tile_kernel<1><<<grid, 256, 0, stream>>>(fiH, fiL, ftH, ftL, negbits,
                                             S_i2t, S_t2i, P_i2t, P_t2i);
    tile_kernel<2><<<grid, 256, 0, stream>>>(fiH, fiL, ftH, ftL, posbits,
                                             S_i2t, S_t2i, P_i2t, P_t2i);

    finalize_kernel<<<1, 256, 0, stream>>>(nneg, npos, P_i2t, P_t2i, out);
}
